// DrugCellAttentionLayer_43035572306419
// MI455X (gfx1250) — matrix operand order repaired and checked by reading, not compiled, not run on hardware
//
#include <hip/hip_runtime.h>

typedef __bf16 bf16_t;
typedef __attribute__((ext_vector_type(16))) __bf16 v16bf;
typedef __attribute__((ext_vector_type(8)))  __bf16 bf16x8;
typedef __attribute__((ext_vector_type(8)))  float   v8f;

namespace {

constexpr int kHid = 512;
constexpr int kHeads = 8;
constexpr int kHd = 64;
constexpr int kB = 32;
constexpr int kT = 256;
constexpr int kS = 512;
constexpr int kPf = 2048;
constexpr float kNeg = -1e10f;

enum { F_RELU = 1, F_OUTF = 2, F_OUTB = 4, F_VT = 8 };

__device__ __forceinline__ v8f wmma_bf16(v16bf a, v16bf b, v8f c) {
  return __builtin_amdgcn_wmma_f32_16x16x32_bf16(false, a, false, b, (short)0, c,
                                                 false, false);
}

// A-fragment: 16x32 bf16, M rows striped lane&15, K split per ISA layout:
// lanes 0-15 hold K=[0..7]+[16..23], lanes 16-31 hold K=[8..15]+[24..31].
__device__ __forceinline__ v16bf frag_a(const bf16_t* __restrict__ A, int lda,
                                        int mBase, int kBase, int lane) {
  int m = mBase + (lane & 15);
  int k = kBase + ((lane >> 4) << 3);
  const bf16_t* p = A + (size_t)m * lda + k;
  bf16x8 lo = *reinterpret_cast<const bf16x8*>(p);
  bf16x8 hi = *reinterpret_cast<const bf16x8*>(p + 16);
  return __builtin_shufflevector(lo, hi, 0, 1, 2, 3, 4, 5, 6, 7, 8, 9, 10, 11,
                                 12, 13, 14, 15);
}

// B-fragment: 32x16 bf16 from K-contiguous rows of Bm[N][ldb]:
// lanes 0-15 hold K=[0..7]+[16..23] of column (lane&15); lanes 16-31 hold K=[8..15]+[24..31].
__device__ __forceinline__ v16bf frag_b(const bf16_t* __restrict__ Bm, int ldb,
                                        int nBase, int kBase, int lane) {
  int n = nBase + (lane & 15);
  int k = kBase + ((lane >> 4) << 3);
  const bf16_t* p = Bm + (size_t)n * ldb + k;
  bf16x8 lo = *reinterpret_cast<const bf16x8*>(p);
  bf16x8 hi = *reinterpret_cast<const bf16x8*>(p + 16);
  return __builtin_shufflevector(lo, hi, 0, 1, 2, 3, 4, 5, 6, 7, 8, 9, 10, 11,
                                 12, 13, 14, 15);
}

// C = act(A[M,K] @ Wt[N,K]^T + bias), epilogue selects f32 / bf16 / per-head
// transposed bf16 (for V). 4 waves per block, each wave: 16(M) x 64(N).
__global__ void __launch_bounds__(128)
gemm_bias_kernel(const bf16_t* __restrict__ A, const bf16_t* __restrict__ Wt,
                 const float* __restrict__ bias, float* __restrict__ outF,
                 bf16_t* __restrict__ outB, int M, int N, int K, int obStride,
                 int obColOff, int flags, int seqLen) {
  int lane = threadIdx.x & 31;
  int wave = threadIdx.x >> 5;
  int mBase = (blockIdx.y * 4 + wave) * 16;
  int nBase = blockIdx.x * 64;
  if (mBase >= M) return;
  v8f acc[4] = {};
  for (int k = 0; k < K; k += 32) {
    v16bf a = frag_a(A, K, mBase, k, lane);
#pragma unroll
    for (int j = 0; j < 4; ++j)
      acc[j] = wmma_bf16(a, frag_b(Wt, K, nBase + j * 16, k, lane), acc[j]);
  }
  int nl = lane & 15;
  int mo = (lane >> 4) * 8;
#pragma unroll
  for (int j = 0; j < 4; ++j) {
    int n = nBase + j * 16 + nl;
    float bv = bias ? bias[n] : 0.0f;
#pragma unroll
    for (int v = 0; v < 8; ++v) {
      int m = mBase + mo + v;
      float val = acc[j][v] + bv;
      if (flags & F_RELU) val = fmaxf(val, 0.0f);
      if (flags & F_OUTF) outF[(size_t)m * N + n] = val;
      if (flags & F_OUTB)
        outB[(size_t)m * obStride + obColOff + n] = (bf16_t)val;
      if (flags & F_VT) {  // write V as [B, H, 64, seqLen]
        int bb = m / seqLen;
        int ss = m - bb * seqLen;
        int h = n >> 6;
        int d = n & 63;
        outB[((size_t)((bb * kHeads + h) * kHd + d)) * seqLen + ss] =
            (bf16_t)val;
      }
    }
  }
}

// Fused attention: one wave per (b, h, 16-query tile).
// energy (WMMA) -> scale+mask -> softmax in LDS -> P(bf16) @ V (WMMA).
__global__ void __launch_bounds__(32)
attention_kernel(const bf16_t* __restrict__ Q, const bf16_t* __restrict__ Kmat,
                 const bf16_t* __restrict__ Vt, const int* __restrict__ mask,
                 bf16_t* __restrict__ Out, int T, int S) {
  extern __shared__ char smem[];
  float* E = reinterpret_cast<float*>(smem);                       // [16][S]
  bf16_t* P = reinterpret_cast<bf16_t*>(smem + (size_t)16 * S * 4);// [16][S+16]
  const int ldp = S + 16;
  int lane = threadIdx.x;
  int qt = blockIdx.x;
  int h = blockIdx.y;
  int b = blockIdx.z;
  int qRow0 = b * T + qt * 16;
  int nl = lane & 15;
  int mo = (lane >> 4) * 8;

  v16bf qa0 = frag_a(Q + h * kHd, kHid, qRow0, 0, lane);
  v16bf qa1 = frag_a(Q + h * kHd, kHid, qRow0, 32, lane);
  const bf16_t* Kbase = Kmat + (size_t)b * S * kHid + h * kHd;
  for (int st = 0; st < S; st += 16) {
    v8f c = {};
    c = wmma_bf16(qa0, frag_b(Kbase, kHid, st, 0, lane), c);
    c = wmma_bf16(qa1, frag_b(Kbase, kHid, st, 32, lane), c);
    int s = st + nl;
    bool dead = (mask[(size_t)b * S + s] == 0);
#pragma unroll
    for (int v = 0; v < 8; ++v)
      E[(size_t)(mo + v) * S + s] = dead ? kNeg : c[v] * 0.125f;
  }
  __syncthreads();
  {  // softmax: two lanes per row, half-row each, combine via shfl_xor(16)
    int r = lane & 15;
    int half = S >> 1;
    int c0 = (lane >> 4) * half;
    float mx = -3.0e38f;
    for (int c = c0; c < c0 + half; ++c) mx = fmaxf(mx, E[(size_t)r * S + c]);
    mx = fmaxf(mx, __shfl_xor(mx, 16, 32));
    float sum = 0.0f;
    for (int c = c0; c < c0 + half; ++c) {
      float e = __expf(E[(size_t)r * S + c] - mx);
      sum += e;
      E[(size_t)r * S + c] = e;
    }
    sum += __shfl_xor(sum, 16, 32);
    float inv = 1.0f / sum;
    for (int c = c0; c < c0 + half; ++c)
      P[(size_t)r * ldp + c] = (bf16_t)(E[(size_t)r * S + c] * inv);
  }
  __syncthreads();
  // P[16,S] @ V[S,64] via Vt rows (K-contiguous in s)
  const bf16_t* Vbase = Vt + (size_t)(b * kHeads + h) * kHd * S;
  v8f acc[4] = {};
  for (int k = 0; k < S; k += 32) {
    int mA = lane & 15;
    int kb = k + ((lane >> 4) << 3);
    const bf16_t* pp = P + (size_t)mA * ldp + kb;
    bf16x8 lo = *reinterpret_cast<const bf16x8*>(pp);
    bf16x8 hi = *reinterpret_cast<const bf16x8*>(pp + 16);
    v16bf a = __builtin_shufflevector(lo, hi, 0, 1, 2, 3, 4, 5, 6, 7, 8, 9, 10,
                                      11, 12, 13, 14, 15);
#pragma unroll
    for (int j = 0; j < 4; ++j)
      acc[j] = wmma_bf16(a, frag_b(Vbase, S, j * 16, k, lane), acc[j]);
  }
#pragma unroll
  for (int j = 0; j < 4; ++j)
#pragma unroll
    for (int v = 0; v < 8; ++v)
      Out[(size_t)(qRow0 + mo + v) * kHid + h * kHd + j * 16 + nl] =
          (bf16_t)acc[j][v];
}

// out = LN(X + Y) * g + b ; optional bf16 copy. One wave32 per row of 512.
__global__ void __launch_bounds__(256)
ln_residual_kernel(const float* __restrict__ X, const float* __restrict__ Y,
                   const float* __restrict__ g, const float* __restrict__ bb,
                   float* __restrict__ outF, bf16_t* __restrict__ outB,
                   int writeB, int rows) {
  int row = blockIdx.x * 8 + (threadIdx.x >> 5);
  if (row >= rows) return;
  int lane = threadIdx.x & 31;
  const float* x = X + (size_t)row * kHid;
  const float* y = Y + (size_t)row * kHid;
  float vals[16];
  float s = 0.0f;
#pragma unroll
  for (int i = 0; i < 16; ++i) {
    int c = lane + i * 32;
    vals[i] = x[c] + y[c];
    s += vals[i];
  }
#pragma unroll
  for (int o = 16; o > 0; o >>= 1) s += __shfl_xor(s, o, 32);
  float mu = s * (1.0f / kHid);
  float vs = 0.0f;
#pragma unroll
  for (int i = 0; i < 16; ++i) {
    float d = vals[i] - mu;
    vs += d * d;
  }
#pragma unroll
  for (int o = 16; o > 0; o >>= 1) vs += __shfl_xor(vs, o, 32);
  float rstd = rsqrtf(vs * (1.0f / kHid) + 1e-5f);
#pragma unroll
  for (int i = 0; i < 16; ++i) {
    int c = lane + i * 32;
    float o = (vals[i] - mu) * rstd * g[c] + bb[c];
    outF[(size_t)row * kHid + c] = o;
    if (writeB) outB[(size_t)row * kHid + c] = (bf16_t)o;
  }
}

__global__ void __launch_bounds__(256)
cvt_bf16_kernel(const float* __restrict__ in, bf16_t* __restrict__ out, long n) {
  long i = (long)blockIdx.x * blockDim.x + threadIdx.x;
  if (i < n) out[i] = (bf16_t)in[i];
}

// W[K,N] f32 row-major -> Wt[N,K] bf16 row-major (K-contiguous B operand)
__global__ void __launch_bounds__(256)
transpose_bf16_kernel(const float* __restrict__ W, bf16_t* __restrict__ Wt,
                      int K, int N) {
  long i = (long)blockIdx.x * blockDim.x + threadIdx.x;
  if (i < (long)K * N) {
    int k = (int)(i / N);
    int n = (int)(i - (long)k * N);
    Wt[(size_t)n * K + k] = (bf16_t)W[i];
  }
}

}  // namespace

extern "C" void kernel_launch(void* const* d_in, const int* in_sizes, int n_in,
                              void* d_out, int out_size, void* d_ws,
                              size_t ws_size, hipStream_t stream) {
  (void)in_sizes; (void)n_in; (void)out_size; (void)ws_size;
  const float* trg = (const float*)d_in[0];
  const int* trg_mask = (const int*)d_in[2];
  const int* src_mask = (const int*)d_in[3];
  const float* ln_g = (const float*)d_in[26];
  const float* ln_b = (const float*)d_in[27];

  char* ws = (char*)d_ws;
  size_t off = 0;
  auto arena = [&](size_t bytes) -> char* {
    char* p = ws + off;
    off += (bytes + 255) & ~(size_t)255;
    return p;
  };
  const int MT = kB * kT;   // 8192
  const int MS = kB * kS;   // 16384

  bf16_t* trgB = (bf16_t*)arena((size_t)MT * kHid * 2);
  bf16_t* encB = (bf16_t*)arena((size_t)MS * kHid * 2);
  // Wt[0..7]=enc q,k,v,o ; self q,k,v,o ; [8]=fc [9]=ff1 [10]=ff2
  const int wKd[11] = {512, 512, 512, 512, 512, 512, 512, 512, 1024, 512, 2048};
  const int wNd[11] = {512, 512, 512, 512, 512, 512, 512, 512, 512, 2048, 512};
  const int wIdx[11] = {4, 6, 8, 10, 12, 14, 16, 18, 20, 22, 24};
  bf16_t* Wt[11];
  for (int i = 0; i < 11; ++i)
    Wt[i] = (bf16_t*)arena((size_t)wKd[i] * wNd[i] * 2);
  bf16_t* Qe = (bf16_t*)arena((size_t)MT * kHid * 2);
  bf16_t* Ke = (bf16_t*)arena((size_t)MS * kHid * 2);
  bf16_t* Vte = (bf16_t*)arena((size_t)MS * kHid * 2);  // [B,H,64,S]
  bf16_t* Qs = (bf16_t*)arena((size_t)MT * kHid * 2);
  bf16_t* Ks = (bf16_t*)arena((size_t)MT * kHid * 2);
  bf16_t* Vts = (bf16_t*)arena((size_t)MT * kHid * 2);  // [B,H,64,T]
  bf16_t* Ae = (bf16_t*)arena((size_t)MT * kHid * 2);
  bf16_t* As = (bf16_t*)arena((size_t)MT * kHid * 2);
  // Lifetime-safe reuse (stream-serialized):
  bf16_t* concatB = encB;        // [MT,1024] bf16 == encB bytes
  float* tBuf = (float*)Ke;      // [MT,512] f32 == Ke bytes
  float* trg1F = (float*)Vte;    // [MT,512] f32
  bf16_t* trg1B = Qe;            // [MT,512] bf16
  bf16_t* ffB = Qs;              // [MT,2048] bf16 spans Qs..Ae
  float* ff2F = (float*)encB;    // [MT,512] f32 (after concatB dead)

  // 1. activations -> bf16
  {
    long n = (long)MT * kHid;
    cvt_bf16_kernel<<<(unsigned)((n + 255) / 256), 256, 0, stream>>>(trg, trgB, n);
    long n2 = (long)MS * kHid;
    cvt_bf16_kernel<<<(unsigned)((n2 + 255) / 256), 256, 0, stream>>>(
        (const float*)d_in[1], encB, n2);
  }
  // 2. weights: transpose + convert
  for (int i = 0; i < 11; ++i) {
    long n = (long)wKd[i] * wNd[i];
    transpose_bf16_kernel<<<(unsigned)((n + 255) / 256), 256, 0, stream>>>(
        (const float*)d_in[wIdx[i]], Wt[i], wKd[i], wNd[i]);
  }
  auto gemm = [&](const bf16_t* A, const bf16_t* W, const float* bias,
                  float* oF, bf16_t* oB, int M, int N, int K, int obStride,
                  int obColOff, int flags, int seqLen) {
    gemm_bias_kernel<<<dim3(N / 64, M / 64), 128, 0, stream>>>(
        A, W, bias, oF, oB, M, N, K, obStride, obColOff, flags, seqLen);
  };
  // 3. Q/K/V projections (bf16 outputs; V transposed per head)
  gemm(trgB, Wt[0], (const float*)d_in[5], nullptr, Qe, MT, 512, 512, 512, 0, F_OUTB, 0);
  gemm(encB, Wt[1], (const float*)d_in[7], nullptr, Ke, MS, 512, 512, 512, 0, F_OUTB, 0);
  gemm(encB, Wt[2], (const float*)d_in[9], nullptr, Vte, MS, 512, 512, 0, 0, F_VT, kS);
  gemm(trgB, Wt[4], (const float*)d_in[13], nullptr, Qs, MT, 512, 512, 512, 0, F_OUTB, 0);
  gemm(trgB, Wt[5], (const float*)d_in[15], nullptr, Ks, MT, 512, 512, 512, 0, F_OUTB, 0);
  gemm(trgB, Wt[6], (const float*)d_in[17], nullptr, Vts, MT, 512, 512, 0, 0, F_VT, kT);
  // 4. fused attention
  {
    dim3 agrid(kT / 16, kHeads, kB);
    size_t smemE = (size_t)16 * kS * 4 + (size_t)16 * (kS + 16) * 2;  // ~49.7KB
    attention_kernel<<<agrid, 32, smemE, stream>>>(Qe, Ke, Vte, src_mask, Ae, kT, kS);
    size_t smemS = (size_t)16 * kT * 4 + (size_t)16 * (kT + 16) * 2;  // ~25KB
    attention_kernel<<<agrid, 32, smemS, stream>>>(Qs, Ks, Vts, trg_mask, As, kT, kT);
  }
  // 5. output projections -> concat buffer [MT,1024] (enc cols 0..511, self 512..1023)
  gemm(Ae, Wt[3], (const float*)d_in[11], nullptr, concatB, MT, 512, 512, 1024, 0, F_OUTB, 0);
  gemm(As, Wt[7], (const float*)d_in[19], nullptr, concatB, MT, 512, 512, 1024, 512, F_OUTB, 0);
  // 6. fc + relu (f32)
  gemm(concatB, Wt[8], (const float*)d_in[21], tBuf, nullptr, MT, 512, 1024, 0, 0, F_OUTF | F_RELU, 0);
  // 7. LN1: trg + t -> trg1 (f32 + bf16)
  ln_residual_kernel<<<MT / 8, 256, 0, stream>>>(trg, tBuf, ln_g, ln_b, trg1F, trg1B, 1, MT);
  // 8. ff1 + relu (bf16)
  gemm(trg1B, Wt[9], (const float*)d_in[23], nullptr, ffB, MT, kPf, 512, kPf, 0, F_OUTB | F_RELU, 0);
  // 9. ff2 (f32)
  gemm(ffB, Wt[10], (const float*)d_in[25], ff2F, nullptr, MT, 512, kPf, 0, 0, F_OUTF, 0);
  // 10. LN2 -> d_out
  ln_residual_kernel<<<MT / 8, 256, 0, stream>>>(trg1F, ff2F, ln_g, ln_b, (float*)d_out, nullptr, 0, MT);
}
